// MultiHeadAttention_40956808135397
// MI455X (gfx1250) — matrix operand order repaired and checked by reading, not compiled, not run on hardware
//
#include <hip/hip_runtime.h>
#include <hip/hip_bf16.h>

typedef __bf16 bf16_t;
typedef __bf16 v16bf __attribute__((ext_vector_type(16)));
typedef __bf16 v8bf  __attribute__((ext_vector_type(8)));
typedef float  v8f   __attribute__((ext_vector_type(8)));

#define D_MODEL 1024
#define NHEAD   16
#define DK      64
#define BATCH   2
#define SEQ     2048
#define M_TOTAL (BATCH * SEQ)   // 4096

union V16U { v16bf v; v8bf h[2]; };

// ---------------------------------------------------------------------------
// fp32 -> bf16 elementwise convert
// ---------------------------------------------------------------------------
__global__ void f32_to_bf16_kernel(const float* __restrict__ in,
                                   bf16_t* __restrict__ out, int n) {
  int i = blockIdx.x * blockDim.x + threadIdx.x;
  int stride = gridDim.x * blockDim.x;
  for (; i < n; i += stride) out[i] = (bf16_t)in[i];
}

// ---------------------------------------------------------------------------
// Wave-level GEMM: Y[m,n] = sum_k A[m,k] * W[n,k] + bias[n]
// A: [M_TOTAL][D_MODEL] bf16, W: [D_MODEL][D_MODEL] bf16 (row = out feature).
// Each wave computes a 16(M) x 128(N) tile: 8 accumulators, K loop step 32
// (8 WMMAs amortize each A-fragment load).
// MODE 0: out bf16 head layout [B][H][S][DK]      (Q, K)
// MODE 1: out bf16 transposed  [B][H][DK][S]      (V)
// MODE 2: out fp32 [M][D_MODEL]                   (final projection)
// ---------------------------------------------------------------------------
template <int MODE>
__global__ __launch_bounds__(256)
void proj_gemm_kernel(const bf16_t* __restrict__ A,
                      const bf16_t* __restrict__ W,
                      const float*  __restrict__ bias,
                      void* __restrict__ Out) {
  int wave = blockIdx.x * (blockDim.x >> 5) + (threadIdx.x >> 5);
  int lane = threadIdx.x & 31;
  int nn = lane & 15;          // N (and row index for A/B lane mapping)
  int hi = lane >> 4;          // which 16-lane half

  int mt = wave >> 3;          // 0..255  -> m0 = mt*16
  int ng = wave & 7;           // 0..7    -> n0 = ng*128
  int m0 = mt * 16;
  int n0 = ng * 128;

  v8f acc[8] = {};

  const bf16_t* arow = A + (size_t)(m0 + nn) * D_MODEL;

  for (int k0 = 0; k0 < D_MODEL; k0 += 32) {
    // A fragment 16x32: lane<16 K={k0..k0+15};
    //                   lane>=16 K={k0+16..k0+31}
    V16U ua;
    const bf16_t* ap = arow + k0 + (hi ? 16 : 0);
    ua.h[0] = *(const v8bf*)(ap);
    ua.h[1] = *(const v8bf*)(ap + 8);

#pragma unroll
    for (int t = 0; t < 8; ++t) {
      // B fragment 32x16: lane holds 16 contiguous K values of column n
      int col = n0 + t * 16 + nn;
      const bf16_t* bp = W + (size_t)col * D_MODEL + k0 + (hi ? 16 : 0);
      v16bf bv = *(const v16bf*)bp;
      acc[t] = __builtin_amdgcn_wmma_f32_16x16x32_bf16(
          false, ua.v, false, bv, (short)0, acc[t], false, false);
    }
  }

  // C/D layout: lane<16 -> (M=r, N=lane); lane>=16 -> (M=r+8, N=lane-16)
  int b = m0 / SEQ;  // constant across tile (SEQ % 16 == 0)
#pragma unroll
  for (int t = 0; t < 8; ++t) {
    int col = n0 + t * 16 + nn;
    float bb = bias[col];
#pragma unroll
    for (int r = 0; r < 8; ++r) {
      int m = m0 + r + (hi ? 8 : 0);
      int s = m - b * SEQ;
      float v = acc[t][r] + bb;
      if (MODE == 0) {
        int h = col >> 6, d = col & 63;
        ((bf16_t*)Out)[(((size_t)(b * NHEAD + h) * SEQ) + s) * DK + d] = (bf16_t)v;
      } else if (MODE == 1) {
        int h = col >> 6, d = col & 63;
        ((bf16_t*)Out)[(((size_t)(b * NHEAD + h) * DK) + d) * SEQ + s] = (bf16_t)v;
      } else {
        ((float*)Out)[(size_t)m * D_MODEL + col] = v;
      }
    }
  }
}

// ---------------------------------------------------------------------------
// Flash attention: one wave per (b,h,q-tile of 16). Streams 64-key tiles
// (4 score sub-tiles folded into each online-softmax row reduction, so the
// cross-lane bpermute/exp cost is amortized over 16 WMMAs per iteration).
// Qh,Kh: [B][H][S][DK] bf16; Vt: [B][H][DK][S] bf16; Ctx: [B][S][D_MODEL] bf16
// ---------------------------------------------------------------------------
__global__ __launch_bounds__(32)
void attn_kernel(const bf16_t* __restrict__ Qh, const bf16_t* __restrict__ Kh,
                 const bf16_t* __restrict__ Vt, bf16_t* __restrict__ Ctx) {
  __shared__ bf16_t pld[16 * 64];

  int lane = threadIdx.x & 31;
  int nn = lane & 15;
  int hi = lane >> 4;

  int blk = blockIdx.x;      // 0 .. B*H*(S/16)-1
  int bh = blk >> 7;         // 0..31
  int qt = blk & 127;
  int q0 = qt * 16;
  int b = bh >> 4, h = bh & 15;

  const bf16_t* qbase = Qh + ((size_t)bh * SEQ + q0) * DK;
  const bf16_t* kbase = Kh + (size_t)bh * SEQ * DK;
  const bf16_t* vbase = Vt + (size_t)bh * DK * SEQ;

  // Q A-fragments for the two d-steps (d 0..31, 32..63)
  v16bf QA[2];
#pragma unroll
  for (int ks = 0; ks < 2; ++ks) {
    V16U u;
    const bf16_t* p = qbase + nn * DK + ks * 32 + (hi ? 16 : 0);
    u.h[0] = *(const v8bf*)p;
    u.h[1] = *(const v8bf*)(p + 8);
    QA[ks] = u.v;
  }

  v8f acc[4] = {};
  float mrow[8], lrow[8];
#pragma unroll
  for (int r = 0; r < 8; ++r) { mrow[r] = -1e30f; lrow[r] = 0.0f; }

  for (int j = 0; j < SEQ / 64; ++j) {
    int key0 = j * 64;

    // Scores: four 16x16 N-subtiles, contraction over d (2 steps of 32)
    v8f sc[4] = {};
#pragma unroll
    for (int sub = 0; sub < 4; ++sub) {
#pragma unroll
      for (int ks = 0; ks < 2; ++ks) {
        const bf16_t* p =
            kbase + (size_t)(key0 + sub * 16 + nn) * DK + ks * 32 + (hi ? 16 : 0);
        v16bf kb = *(const v16bf*)p;
        sc[sub] = __builtin_amdgcn_wmma_f32_16x16x32_bf16(
            false, QA[ks], false, kb, (short)0, sc[sub], false, false);
      }
    }

    // Online softmax; P written straight to LDS in [16][64] bf16
    __syncthreads();
#pragma unroll
    for (int r = 0; r < 8; ++r) {
      float s0 = sc[0][r] * 0.125f;   // 1/sqrt(64)
      float s1 = sc[1][r] * 0.125f;
      float s2 = sc[2][r] * 0.125f;
      float s3 = sc[3][r] * 0.125f;
      float mt = fmaxf(fmaxf(s0, s1), fmaxf(s2, s3));
#pragma unroll
      for (int msk = 1; msk < 16; msk <<= 1)
        mt = fmaxf(mt, __shfl_xor(mt, msk, 32));
      float mnew = fmaxf(mrow[r], mt);
      float alpha = __expf(mrow[r] - mnew);
      float e0 = __expf(s0 - mnew);
      float e1 = __expf(s1 - mnew);
      float e2 = __expf(s2 - mnew);
      float e3 = __expf(s3 - mnew);
      float rs = (e0 + e1) + (e2 + e3);
#pragma unroll
      for (int msk = 1; msk < 16; msk <<= 1)
        rs += __shfl_xor(rs, msk, 32);
      lrow[r] = lrow[r] * alpha + rs;
      mrow[r] = mnew;
#pragma unroll
      for (int t = 0; t < 4; ++t) acc[t][r] *= alpha;

      int row = r + (hi ? 8 : 0);
      pld[row * 64 + nn]      = (bf16_t)e0;
      pld[row * 64 + 16 + nn] = (bf16_t)e1;
      pld[row * 64 + 32 + nn] = (bf16_t)e2;
      pld[row * 64 + 48 + nn] = (bf16_t)e3;
    }
    __syncthreads();

    // P A-fragments for the two key-chunks (keys 0..31, 32..63 of the tile)
    v16bf PA[2];
#pragma unroll
    for (int kc = 0; kc < 2; ++kc) {
      V16U u;
      const bf16_t* p = pld + nn * 64 + kc * 32 + (hi ? 16 : 0);
      u.h[0] = *(const v8bf*)p;
      u.h[1] = *(const v8bf*)(p + 8);
      PA[kc] = u.v;
    }

    // out += P @ V : B lane = output-d column, 16 contiguous keys from Vt
#pragma unroll
    for (int t = 0; t < 4; ++t) {
#pragma unroll
      for (int kc = 0; kc < 2; ++kc) {
        const bf16_t* p =
            vbase + (size_t)(t * 16 + nn) * SEQ + key0 + kc * 32 + (hi ? 16 : 0);
        v16bf vb = *(const v16bf*)p;
        acc[t] = __builtin_amdgcn_wmma_f32_16x16x32_bf16(
            false, PA[kc], false, vb, (short)0, acc[t], false, false);
      }
    }
  }

  // Normalize and write context in [B][S][D_MODEL] bf16
#pragma unroll
  for (int r = 0; r < 8; ++r) {
    float inv = 1.0f / lrow[r];
    int s = q0 + r + (hi ? 8 : 0);
#pragma unroll
    for (int t = 0; t < 4; ++t) {
      int col = h * DK + t * 16 + nn;
      Ctx[((size_t)b * SEQ + s) * D_MODEL + col] = (bf16_t)(acc[t][r] * inv);
    }
  }
}

// ---------------------------------------------------------------------------
// Launch
// ---------------------------------------------------------------------------
extern "C" void kernel_launch(void* const* d_in, const int* in_sizes, int n_in,
                              void* d_out, int out_size, void* d_ws, size_t ws_size,
                              hipStream_t stream) {
  const float* q  = (const float*)d_in[0];
  const float* k  = (const float*)d_in[1];
  const float* v  = (const float*)d_in[2];
  const float* Wq = (const float*)d_in[3];
  const float* bq = (const float*)d_in[4];
  const float* Wk = (const float*)d_in[5];
  const float* bk = (const float*)d_in[6];
  const float* Wv = (const float*)d_in[7];
  const float* bv = (const float*)d_in[8];
  const float* Wo = (const float*)d_in[9];
  const float* bo = (const float*)d_in[10];

  const size_t MB = 1u << 20;
  char* ws = (char*)d_ws;
  bf16_t* qb  = (bf16_t*)(ws + 0 * MB);    // 8 MB each
  bf16_t* kb  = (bf16_t*)(ws + 8 * MB);
  bf16_t* vb  = (bf16_t*)(ws + 16 * MB);
  bf16_t* wqb = (bf16_t*)(ws + 24 * MB);   // 2 MB each
  bf16_t* wkb = (bf16_t*)(ws + 26 * MB);
  bf16_t* wvb = (bf16_t*)(ws + 28 * MB);
  bf16_t* wob = (bf16_t*)(ws + 30 * MB);
  bf16_t* Qh  = (bf16_t*)(ws + 32 * MB);   // 8 MB each
  bf16_t* Kh  = (bf16_t*)(ws + 40 * MB);
  bf16_t* Vt  = (bf16_t*)(ws + 48 * MB);
  bf16_t* Ctx = (bf16_t*)(ws + 56 * MB);

  const int N_ACT = M_TOTAL * D_MODEL;     // 4194304
  const int N_W   = D_MODEL * D_MODEL;     // 1048576

  f32_to_bf16_kernel<<<1024, 256, 0, stream>>>(q,  qb,  N_ACT);
  f32_to_bf16_kernel<<<1024, 256, 0, stream>>>(k,  kb,  N_ACT);
  f32_to_bf16_kernel<<<1024, 256, 0, stream>>>(v,  vb,  N_ACT);
  f32_to_bf16_kernel<<<512,  256, 0, stream>>>(Wq, wqb, N_W);
  f32_to_bf16_kernel<<<512,  256, 0, stream>>>(Wk, wkb, N_W);
  f32_to_bf16_kernel<<<512,  256, 0, stream>>>(Wv, wvb, N_W);
  f32_to_bf16_kernel<<<512,  256, 0, stream>>>(Wo, wob, N_W);

  // 2048 waves = 256 M-tiles x 8 N-groups (16x128 tile); 8 waves per block
  proj_gemm_kernel<0><<<256, 256, 0, stream>>>(qb, wqb, bq, (void*)Qh);
  proj_gemm_kernel<0><<<256, 256, 0, stream>>>(kb, wkb, bk, (void*)Kh);
  proj_gemm_kernel<1><<<256, 256, 0, stream>>>(vb, wvb, bv, (void*)Vt);

  // one wave per (b,h,16-query tile): 2*16*128 = 4096 blocks
  attn_kernel<<<4096, 32, 0, stream>>>(Qh, Kh, Vt, Ctx);

  proj_gemm_kernel<2><<<256, 256, 0, stream>>>(Ctx, wob, bo, d_out);
}
